// ODEBlock_3427383902476
// MI455X (gfx1250) — matrix operand order repaired and checked by reading, not compiled, not run on hardware
//
#include <hip/hip_runtime.h>
#include <math.h>
#include <stdint.h>

// ---------------------------------------------------------------------------
// Neural ODE block on gfx1250: fixed-step RK4, f(z) = tanh(z@W1+b1)@W2+b2.
// GEMMs: v_wmma_f32_16x16x32_bf16 with 2-way split precision (bf16x3):
//   x = hi + lo (bf16);  A@B ~= AhiBhi + AhiBlo + AloBhi  (f32 accumulate)
// Weights pre-split once into bf16 hi/lo, K-major; B tiles staged with
// global_load_async_to_lds_b128 (ASYNCcnt) into double-buffered LDS; A tiles
// register-pipelined: global_load_b128 for chunk kc+32 issued during the
// WMMA phase of chunk kc, converted + ds_store_b128'd next iteration.
// ---------------------------------------------------------------------------

typedef __attribute__((ext_vector_type(16))) __bf16 v16bf;
typedef __attribute__((ext_vector_type(8)))  float  v8f;

#define BM   128
#define BN   128
#define BK   32
#define LDSA 40   // A tile row stride (ushort): 80B, 16B-aligned, 16 bank groups
#define LDSB 40   // B tile row stride (ushort): 80B, 16B-aligned for async B128

__device__ __forceinline__ unsigned short f32_to_bf16(float f) {
    unsigned int u = __float_as_uint(f);
    u += 0x7FFFu + ((u >> 16) & 1u);          // round-to-nearest-even
    return (unsigned short)(u >> 16);
}
__device__ __forceinline__ float bf16_to_f32(unsigned short h) {
    return __uint_as_float(((unsigned int)h) << 16);
}

// gfx1250 async copy: 16B per lane, global -> LDS, tracked by ASYNCcnt.
__device__ __forceinline__ void async_b128_to_lds(unsigned lds_byte_addr,
                                                  const void* sbase,
                                                  unsigned voffset_bytes) {
    asm volatile("global_load_async_to_lds_b128 %0, %1, %2"
                 :
                 : "v"(lds_byte_addr), "v"(voffset_bytes), "s"(sbase)
                 : "memory");
}

union Frag { v16bf v; unsigned int u[8]; };
union Acc8 { v8f v; float f[8]; };

// One-time: split f32 W[K][N] into bf16 hi/lo stored K-major ([N][K]).
__global__ __launch_bounds__(256)
void neuralode_split_w(const float* __restrict__ W,
                       unsigned short* __restrict__ Whi,
                       unsigned short* __restrict__ Wlo,
                       int K, int N)
{
    const int idx = blockIdx.x * blockDim.x + threadIdx.x;
    if (idx < K * N) {
        const int k = idx / N;
        const int n = idx - k * N;
        const float x = W[idx];
        const unsigned short h = f32_to_bf16(x);
        const unsigned short l = f32_to_bf16(x - bf16_to_f32(h));
        Whi[(size_t)n * K + k] = h;
        Wlo[(size_t)n * K + k] = l;
    }
}

// GEMM:  Out = epilogue( (A0 + aCoef*Aadd) @ W + bias ), W given pre-split.
//   applyTanh==1 : Out = tanh(.)                       (layer-1 path, H buffer)
//   applyTanh==0 : Out = (.) ;  Acc = beta*Acc + w*(.) (layer-2 path, k buffer)
__global__ __launch_bounds__(256)
void neuralode_gemm_bf16x3(const float* __restrict__ A0,
                           const float* __restrict__ Aadd, float aCoef,
                           const unsigned short* __restrict__ Whi,  // [N][K]
                           const unsigned short* __restrict__ Wlo,  // [N][K]
                           const float* __restrict__ bias,
                           float* __restrict__ Out,
                           float* __restrict__ Acc,
                           float wCoef, int betaFlag, int applyTanh,
                           int M, int N, int K)
{
    __shared__ __align__(16) unsigned short sAhi[BM * LDSA], sAlo[BM * LDSA];
    __shared__ __align__(16) unsigned short sBhi[2][BN * LDSB];
    __shared__ __align__(16) unsigned short sBlo[2][BN * LDSB];

    const int t    = threadIdx.x;
    const int lane = t & 31;               // wave32
    const int wv   = t >> 5;               // 8 waves / block
    const int wm   = wv & 3;               // wave M tile: wm*32
    const int wn   = wv >> 2;              // wave N tile: wn*64
    const int blockM = blockIdx.y * BM;
    const int blockN = blockIdx.x * BN;
    const bool haveAdd = (aCoef != 0.f);   // uniform

    Acc8 acc[2][4];
    #pragma unroll
    for (int i = 0; i < 2; ++i)
        #pragma unroll
        for (int j = 0; j < 4; ++j)
            #pragma unroll
            for (int e = 0; e < 8; ++e) acc[i][j].f[e] = 0.f;

    const int aRow  = t >> 1;              // 0..127 (2 threads/row, 16 cols each)
    const int aCol0 = (t & 1) * 16;
    const size_t aBaseRow = (size_t)(blockM + aRow) * K + aCol0;

    // ---- register-pipelined A staging ----
    float4 zr[4], kr[4];
    auto loadA = [&](int kcc) {
        const float4* pz = (const float4*)(A0 + aBaseRow + kcc);
        #pragma unroll
        for (int j = 0; j < 4; ++j) zr[j] = pz[j];
        if (haveAdd) {
            const float4* pk = (const float4*)(Aadd + aBaseRow + kcc);
            #pragma unroll
            for (int j = 0; j < 4; ++j) kr[j] = pk[j];
        }
    };
    auto storeA = [&]() {
        float x[16];
        #pragma unroll
        for (int j = 0; j < 4; ++j) {
            x[4*j+0] = zr[j].x; x[4*j+1] = zr[j].y;
            x[4*j+2] = zr[j].z; x[4*j+3] = zr[j].w;
        }
        if (haveAdd) {
            #pragma unroll
            for (int j = 0; j < 4; ++j) {
                x[4*j+0] = fmaf(aCoef, kr[j].x, x[4*j+0]);
                x[4*j+1] = fmaf(aCoef, kr[j].y, x[4*j+1]);
                x[4*j+2] = fmaf(aCoef, kr[j].z, x[4*j+2]);
                x[4*j+3] = fmaf(aCoef, kr[j].w, x[4*j+3]);
            }
        }
        unsigned hi[8], lo[8];
        #pragma unroll
        for (int j = 0; j < 8; ++j) {
            const unsigned short h0 = f32_to_bf16(x[2*j]);
            const unsigned short h1 = f32_to_bf16(x[2*j+1]);
            const unsigned short l0 = f32_to_bf16(x[2*j]   - bf16_to_f32(h0));
            const unsigned short l1 = f32_to_bf16(x[2*j+1] - bf16_to_f32(h1));
            hi[j] = (unsigned)h0 | ((unsigned)h1 << 16);
            lo[j] = (unsigned)l0 | ((unsigned)l1 << 16);
        }
        uint4* dHi = (uint4*)&sAhi[aRow * LDSA + aCol0];
        uint4* dLo = (uint4*)&sAlo[aRow * LDSA + aCol0];
        dHi[0] = make_uint4(hi[0], hi[1], hi[2], hi[3]);
        dHi[1] = make_uint4(hi[4], hi[5], hi[6], hi[7]);
        dLo[0] = make_uint4(lo[0], lo[1], lo[2], lo[3]);
        dLo[1] = make_uint4(lo[4], lo[5], lo[6], lo[7]);
    };

    // Per-thread async-copy mapping for B tiles: 512 16B chunks per array,
    // 2 per thread per array -> 4 async issues per thread per buffer.
    const unsigned ldsBhiBase = (unsigned)(uintptr_t)&sBhi[0][0];
    const unsigned ldsBloBase = (unsigned)(uintptr_t)&sBlo[0][0];
    const unsigned bufStride  = (unsigned)(BN * LDSB * sizeof(unsigned short));
    auto prefetchB = [&](int nbuf, int kcN) {
        #pragma unroll
        for (int q = 0; q < 2; ++q) {
            const int c    = t + 256 * q;         // 0..511
            const int n    = c >> 2;              // row within tile (0..127)
            const int part = c & 3;               // 16B sub-chunk
            const unsigned ldsOff = (unsigned)(n * (LDSB * 2) + part * 16)
                                  + (unsigned)nbuf * bufStride;
            const unsigned goff = (unsigned)(((blockN + n) * K + kcN) * 2 + part * 16);
            async_b128_to_lds(ldsBhiBase + ldsOff, Whi, goff);
            async_b128_to_lds(ldsBloBase + ldsOff, Wlo, goff);
        }
    };

    loadA(0);          // warm up A registers
    prefetchB(0, 0);   // warm up B buffer 0

    int buf = 0;
    for (int kc = 0; kc < K; kc += BK) {
        // ---- commit prefetched A registers to LDS (fused z + a*k, hi/lo) ----
        storeA();

        // ---- prefetch next B chunk into the other buffer, then wait for
        //      the current buffer (async loads retire in order: ASYNCcnt<=4
        //      means the 4 current-buffer issues have landed). ----
        const int kcNext = kc + BK;
        if (kcNext < K) {
            prefetchB(buf ^ 1, kcNext);
            asm volatile("s_wait_asynccnt 0x4" ::: "memory");
        } else {
            asm volatile("s_wait_asynccnt 0x0" ::: "memory");
        }
        __syncthreads();

        // ---- issue next A global loads; they complete during the WMMAs ----
        if (kcNext < K) loadA(kcNext);

        // ---- A fragments (ISA 16-bit A layout: lanes<16 K={0..7,16..23}) ----
        Frag aHi[2], aLo[2];
        const int koffA = (lane < 16) ? 0 : 8;
        #pragma unroll
        for (int ti = 0; ti < 2; ++ti) {
            const int row = wm * 32 + ti * 16 + (lane & 15);
            const unsigned short* pHi = &sAhi[row * LDSA];
            const unsigned short* pLo = &sAlo[row * LDSA];
            #pragma unroll
            for (int p = 0; p < 8; ++p) {
                const int kk = ((p < 4) ? 2 * p : 2 * p + 8) + koffA;
                aHi[ti].u[p] = *(const unsigned int*)(pHi + kk);
                aLo[ti].u[p] = *(const unsigned int*)(pLo + kk);
            }
        }
        // ---- B fragments (lanes<16 hold K={0..7,16..23}; lanes>=16 K={8..15,24..31}) ----
        const int koffB = (lane < 16) ? 0 : 8;
        #pragma unroll
        for (int tj = 0; tj < 4; ++tj) {
            Frag bHi, bLo;
            const int col = wn * 64 + tj * 16 + (lane & 15);
            const unsigned short* pHi = &sBhi[buf][col * LDSB];
            const unsigned short* pLo = &sBlo[buf][col * LDSB];
            #pragma unroll
            for (int p = 0; p < 8; ++p) {
                const int kk = ((p < 4) ? 2 * p : 2 * p + 8) + koffB;
                bHi.u[p] = *(const unsigned int*)(pHi + kk);
                bLo.u[p] = *(const unsigned int*)(pLo + kk);
            }
            #pragma unroll
            for (int ti = 0; ti < 2; ++ti) {
                acc[ti][tj].v = __builtin_amdgcn_wmma_f32_16x16x32_bf16(
                    false, aHi[ti].v, false, bHi.v, (short)0, acc[ti][tj].v, false, false);
                acc[ti][tj].v = __builtin_amdgcn_wmma_f32_16x16x32_bf16(
                    false, aHi[ti].v, false, bLo.v, (short)0, acc[ti][tj].v, false, false);
                acc[ti][tj].v = __builtin_amdgcn_wmma_f32_16x16x32_bf16(
                    false, aLo[ti].v, false, bHi.v, (short)0, acc[ti][tj].v, false, false);
            }
        }
        __syncthreads();
        buf ^= 1;
    }

    // ---- epilogue: C/D layout = VGPR v -> M = v + 8*(lane>=16), N = lane&15 ----
    const int laneN  = lane & 15;
    const int rowOff = (lane >> 4) * 8;
    #pragma unroll
    for (int ti = 0; ti < 2; ++ti) {
        #pragma unroll
        for (int tj = 0; tj < 4; ++tj) {
            const int gN = blockN + wn * 64 + tj * 16 + laneN;
            const float b = bias[gN];
            #pragma unroll
            for (int v = 0; v < 8; ++v) {
                const int gM = blockM + wm * 32 + ti * 16 + rowOff + v;
                const size_t idx = (size_t)gM * N + gN;
                const float val = acc[ti][tj].f[v] + b;
                if (applyTanh) {
                    Out[idx] = tanhf(val);
                } else {
                    Out[idx] = val;                 // k_i for next probe point
                    const float a = wCoef * val;    // RK weight accumulation
                    if (betaFlag) Acc[idx] += a; else Acc[idx] = a;
                }
            }
        }
    }
}

__global__ __launch_bounds__(256)
void neuralode_axpy_update(float* __restrict__ z, const float* __restrict__ acc, int n)
{
    const int i = blockIdx.x * blockDim.x + threadIdx.x;
    if (i < n) z[i] += acc[i];
}

extern "C" void kernel_launch(void* const* d_in, const int* in_sizes, int n_in,
                              void* d_out, int out_size, void* d_ws, size_t ws_size,
                              hipStream_t stream)
{
    (void)in_sizes; (void)n_in; (void)out_size; (void)ws_size;
    const int BATCH = 2048, D = 1024, HD = 1024;

    const float* z0 = (const float*)d_in[0];
    const float* W1 = (const float*)d_in[1];
    const float* b1 = (const float*)d_in[2];
    const float* W2 = (const float*)d_in[3];
    const float* b2 = (const float*)d_in[4];
    float* out = (float*)d_out;

    const size_t ZN = (size_t)BATCH * D;
    const size_t WN = (size_t)D * HD;
    float* ws   = (float*)d_ws;
    float* Zcur = ws;            // current state
    float* Hbuf = ws + ZN;       // hidden activations
    float* Kbuf = ws + 2 * ZN;   // current RK slope k_i
    float* Acc  = ws + 3 * ZN;   // dt/6*(k1+2k2+2k3+k4) running sum
    unsigned short* wsplit = (unsigned short*)(ws + 4 * ZN);
    unsigned short* W1hi = wsplit;            // [HD][D] K-major
    unsigned short* W1lo = wsplit + WN;
    unsigned short* W2hi = wsplit + 2 * WN;   // [D][HD] K-major
    unsigned short* W2lo = wsplit + 3 * WN;

    hipMemcpyAsync(Zcur, z0, ZN * sizeof(float), hipMemcpyDeviceToDevice, stream);

    // One-time bf16 hi/lo split of the (loop-invariant) weights, K-major.
    {
        const int nw = (int)WN;
        const dim3 gs((nw + 255) / 256), bs(256);
        neuralode_split_w<<<gs, bs, 0, stream>>>(W1, W1hi, W1lo, D, HD);
        neuralode_split_w<<<gs, bs, 0, stream>>>(W2, W2hi, W2lo, HD, D);
    }

    const int   NSTEPS = 10;
    const float dt = 1.0f / (float)NSTEPS;
    const float aC[4] = {0.f, 0.5f * dt, 0.5f * dt, dt};          // probe offsets
    const float wC[4] = {dt / 6.f, dt / 3.f, dt / 3.f, dt / 6.f}; // RK4 weights

    const dim3 blk(256);
    const dim3 g1(HD / BN, BATCH / BM);   // GEMM1: (B,D)x(D,H)
    const dim3 g2(D  / BN, BATCH / BM);   // GEMM2: (B,H)x(H,D)
    const int  nUpd = (int)ZN;
    const dim3 gu((nUpd + 255) / 256);

    for (int s = 0; s < NSTEPS; ++s) {
        for (int e = 0; e < 4; ++e) {
            // H = tanh((Zcur + a_e * k_prev) @ W1 + b1)
            neuralode_gemm_bf16x3<<<g1, blk, 0, stream>>>(
                Zcur, Kbuf, aC[e], W1hi, W1lo, b1, Hbuf, nullptr,
                0.f, 0, /*tanh=*/1, BATCH, HD, D);
            // k_e = H @ W2 + b2 ;  Acc = beta*Acc + w_e * k_e
            neuralode_gemm_bf16x3<<<g2, blk, 0, stream>>>(
                Hbuf, nullptr, 0.f, W2hi, W2lo, b2, Kbuf, Acc,
                wC[e], (e == 0) ? 0 : 1, /*tanh=*/0, BATCH, D, HD);
        }
        neuralode_axpy_update<<<gu, blk, 0, stream>>>(Zcur, Acc, nUpd);
    }

    hipMemcpyAsync(out, Zcur, ZN * sizeof(float), hipMemcpyDeviceToDevice, stream);
}
